// BigramLanguageModel_23716809409344
// MI455X (gfx1250) — matrix operand order repaired and checked by reading, not compiled, not run on hardware
//
#include <hip/hip_runtime.h>

// ---------------- problem constants ----------------
#define T_SEQ   16
#define C_EMB   40
#define H_HEADS 4
#define D_HEAD  10
#define VOCAB   50257
#define ROWS    4096              // B*T = 256*16
#define NCHUNK  50                // ceil(50257 / 1024)
#define CHUNK_COLS 1024           // 8 waves * 8 tiles * 16 cols
#define NTILE_B (NCHUNK * 64)     // 3200 padded 16-col tiles
#define FRAG_US 16                // bf16 elements per lane per fragment

typedef __attribute__((ext_vector_type(16))) __bf16 v16bf;
typedef __attribute__((ext_vector_type(8)))  float  v8f;

#define NEG_INF (-__builtin_inff())

static __device__ __forceinline__ unsigned short bf16bits(float f) {
    return __builtin_bit_cast(unsigned short, (__bf16)f);
}

// =====================================================================
// Kernel 1: x = tok_emb[idx] + pos_emb ; 4-head causal attention; then
// pack this block's 16 output rows straight into the WMMA A-fragment
// layout (lane l16 = row M, element e -> K = e + (e>=8?8:0) + 8*hh + 32*cc).
// One workgroup per batch element == one 16-row A tile.
// =====================================================================
__global__ __launch_bounds__(256)
void embed_attn_pack_kernel(const int* __restrict__ idx,
                            const float* __restrict__ tok_emb,
                            const float* __restrict__ pos_emb,
                            const float* __restrict__ Wq,
                            const float* __restrict__ Wk,
                            const float* __restrict__ Wv,
                            unsigned short* __restrict__ xpack, // [256][2][32][16]
                            float* __restrict__ loss_slot)
{
    __shared__ float x[T_SEQ][C_EMB];
    __shared__ float qs[H_HEADS][T_SEQ][D_HEAD];
    __shared__ float ks[H_HEADS][T_SEQ][D_HEAD];
    __shared__ float vs[H_HEADS][T_SEQ][D_HEAD];
    __shared__ float att[H_HEADS][T_SEQ][T_SEQ];
    __shared__ float xo[T_SEQ][C_EMB];

    const int b   = blockIdx.x;
    const int tid = threadIdx.x;
    if (b == 0 && tid == 0) *loss_slot = 0.0f;        // deterministic re-init

    for (int i = tid; i < T_SEQ * C_EMB; i += 256) {
        int t = i / C_EMB, c = i % C_EMB;
        x[t][c] = tok_emb[(size_t)idx[b * T_SEQ + t] * C_EMB + c] + pos_emb[t * C_EMB + c];
    }
    __syncthreads();

    for (int i = tid; i < H_HEADS * T_SEQ * D_HEAD; i += 256) {
        int h = i / (T_SEQ * D_HEAD), r = i % (T_SEQ * D_HEAD);
        int t = r / D_HEAD, d = r % D_HEAD;
        float sq = 0.f, sk = 0.f, sv = 0.f;
        for (int c = 0; c < C_EMB; ++c) {
            float xv = x[t][c];
            sq += xv * Wq[(h * C_EMB + c) * D_HEAD + d];
            sk += xv * Wk[(h * C_EMB + c) * D_HEAD + d];
            sv += xv * Wv[(h * C_EMB + c) * D_HEAD + d];
        }
        qs[h][t][d] = sq; ks[h][t][d] = sk; vs[h][t][d] = sv;
    }
    __syncthreads();

    for (int i = tid; i < H_HEADS * T_SEQ * T_SEQ; i += 256) {
        int h = i / (T_SEQ * T_SEQ), r = i % (T_SEQ * T_SEQ);
        int t = r / T_SEQ, s = r % T_SEQ;
        float a = NEG_INF;
        if (s <= t) {
            a = 0.f;
            for (int d = 0; d < D_HEAD; ++d) a += qs[h][t][d] * ks[h][s][d];
        }
        att[h][t][s] = a;
    }
    __syncthreads();

    if (tid < H_HEADS * T_SEQ) {            // softmax rows (no 1/sqrt(d), faithful)
        int h = tid / T_SEQ, t = tid % T_SEQ;
        float m = NEG_INF;
        for (int s = 0; s <= t; ++s) m = fmaxf(m, att[h][t][s]);
        float sum = 0.f;
        for (int s = 0; s < T_SEQ; ++s) {
            float e = (s <= t) ? __expf(att[h][t][s] - m) : 0.f;
            att[h][t][s] = e; sum += e;
        }
        float inv = 1.f / sum;
        for (int s = 0; s < T_SEQ; ++s) att[h][t][s] *= inv;
    }
    __syncthreads();

    for (int i = tid; i < H_HEADS * T_SEQ * D_HEAD; i += 256) {
        int h = i / (T_SEQ * D_HEAD), r = i % (T_SEQ * D_HEAD);
        int t = r / D_HEAD, d = r % D_HEAD;
        float o = 0.f;
        for (int s = 0; s <= t; ++s) o += att[h][t][s] * vs[h][s][d];
        xo[t][h * D_HEAD + d] = o;
    }
    __syncthreads();

    // ---- pack A fragments: (cc, lane) -> 16 bf16 -> one 32B slot ----
    if (tid < 64) {
        int cc   = tid >> 5;            // K chunk 0/1
        int lane = tid & 31;
        int hh   = lane >> 4;
        int l16  = lane & 15;
        union { unsigned short u[FRAG_US]; uint4 q[2]; } pk;
#pragma unroll
        for (int e = 0; e < FRAG_US; ++e) {
            int k = e + ((e >= 8) ? 8 : 0) + 8 * hh + 32 * cc;
            pk.u[e] = (k < C_EMB) ? bf16bits(xo[l16][k]) : (unsigned short)0;
        }
        uint4* dst = (uint4*)&xpack[((size_t)(b * 2 + cc) * 32 + lane) * FRAG_US];
        dst[0] = pk.q[0];
        dst[1] = pk.q[1];
    }
}

// =====================================================================
// Kernel 1b: pack lm_W into WMMA B-fragment layout, once.
// lane l16 = column N, element e -> K = e + (e>=8?8:0) + 8*hh + 32*cc (K>=40 zeroed).
// [NTILE_B][2][32][16 bf16] = 6.4 MB, stays L2-resident for the GEMM.
// =====================================================================
__global__ __launch_bounds__(256)
void pack_lmW_kernel(const float* __restrict__ lm_W,
                     unsigned short* __restrict__ wpack)
{
    int gid = blockIdx.x * 256 + threadIdx.x;     // [NTILE_B * 2 * 32)
    if (gid >= NTILE_B * 64) return;
    int lane = gid & 31;
    int cc   = (gid >> 5) & 1;
    int tt   = gid >> 6;
    int hh   = lane >> 4;
    int l16  = lane & 15;
    int col  = tt * 16 + l16;
    int ccol = (col < VOCAB) ? col : (VOCAB - 1);   // clamp; masked out later

    union { unsigned short u[FRAG_US]; uint4 q[2]; } pk;
#pragma unroll
    for (int e = 0; e < FRAG_US; ++e) {
        int k = e + ((e >= 8) ? 8 : 0) + 8 * hh + 32 * cc;
        pk.u[e] = (k < C_EMB) ? bf16bits(lm_W[(size_t)k * VOCAB + ccol])
                              : (unsigned short)0;
    }
    uint4* dst = (uint4*)&wpack[(size_t)gid * FRAG_US];
    dst[0] = pk.q[0];
    dst[1] = pk.q[1];
}

// =====================================================================
// Kernel 2: logits = A @ B + b with pre-packed bf16 fragments.
// Per tile: 2x v16bf B loads (b128 pairs), 2 WMMAs (K=40->64), bias add,
// per-lane online softmax stats, 8 NT stores. Cross-lane merge once at end.
// =====================================================================
__global__ __launch_bounds__(256)
void gemm_logits_kernel(const unsigned short* __restrict__ xpack,
                        const unsigned short* __restrict__ wpack,
                        const float* __restrict__ lm_b,
                        float* __restrict__ logits,       // [ROWS][VOCAB]
                        float* __restrict__ chunkMax,     // [ROWS][NCHUNK]
                        float* __restrict__ chunkSum)     // [ROWS][NCHUNK]
{
    const int tid  = threadIdx.x;
    const int lane = tid & 31;
    const int wave = tid >> 5;
    const int hh   = lane >> 4;
    const int l16  = lane & 15;
    const int chunk   = blockIdx.x;
    const int rowTile = blockIdx.y;
    const int rowBase = rowTile * 16;

    // A fragments: one 32B vector load each, reused for all 8 tiles
    const v16bf a0 = *(const v16bf*)&xpack[((size_t)(rowTile * 2 + 0) * 32 + lane) * FRAG_US];
    const v16bf a1 = *(const v16bf*)&xpack[((size_t)(rowTile * 2 + 1) * 32 + lane) * FRAG_US];

    float rmax[8], rsum[8];
#pragma unroll
    for (int v = 0; v < 8; ++v) { rmax[v] = NEG_INF; rsum[v] = 0.f; }

    const int tileBase = chunk * 64 + wave * 8;   // this wave's 8 B tiles

    for (int j = 0; j < 8; ++j) {
        const int  tileIdx = tileBase + j;
        const int  col     = tileIdx * 16 + l16;
        const bool cvalid  = (col < VOCAB);
        const int  ccol    = cvalid ? col : (VOCAB - 1);

        const v16bf b0 = *(const v16bf*)&wpack[((size_t)(tileIdx * 2 + 0) * 32 + lane) * FRAG_US];
        const v16bf b1 = *(const v16bf*)&wpack[((size_t)(tileIdx * 2 + 1) * 32 + lane) * FRAG_US];

        // prefetch next tile's fragments (global_prefetch_b8); wpack is L2-hot
        if (j < 7)
            __builtin_prefetch(&wpack[((size_t)(tileIdx * 2 + 2) * 32 + lane) * FRAG_US], 0, 3);

        const float bias = lm_b[ccol];

        v8f c = {};
        c = __builtin_amdgcn_wmma_f32_16x16x32_bf16(false, a0, false, b0,
                                                    (short)0, c, false, false);
        c = __builtin_amdgcn_wmma_f32_16x16x32_bf16(false, a1, false, b1,
                                                    (short)0, c, false, false);

        // C/D layout: lane l16 = col, VGPR v = row v + 8*hh.
        if (cvalid) {
#pragma unroll
            for (int v = 0; v < 8; ++v) {
                float val = c[v] + bias;
                float nm  = fmaxf(rmax[v], val);
                rsum[v]   = rsum[v] * __expf(rmax[v] - nm) + __expf(val - nm);
                rmax[v]   = nm;
                int grow  = rowBase + v + hh * 8;
                __builtin_nontemporal_store(val, &logits[(size_t)grow * VOCAB + col]);
            }
        }
    }

    // ---- one cross-lane (16-lane half) merge of (max,sum) per row ----
#pragma unroll
    for (int v = 0; v < 8; ++v) {
        float m = rmax[v], s = rsum[v];
#pragma unroll
        for (int mask = 1; mask <= 8; mask <<= 1) {
            float m2 = __shfl_xor(m, mask, 32);
            float s2 = __shfl_xor(s, mask, 32);
            float nm = fmaxf(m, m2);
            float t1 = (m  > NEG_INF) ? s  * __expf(m  - nm) : 0.f;
            float t2 = (m2 > NEG_INF) ? s2 * __expf(m2 - nm) : 0.f;
            m = nm; s = t1 + t2;
        }
        rmax[v] = m; rsum[v] = s;
    }

    // ---- merge 8 waves' per-row stats for this 1024-col chunk ----
    __shared__ float smax[8][16];
    __shared__ float ssum[8][16];
    if (l16 == 0) {
#pragma unroll
        for (int v = 0; v < 8; ++v) {
            smax[wave][v + hh * 8] = rmax[v];
            ssum[wave][v + hh * 8] = rsum[v];
        }
    }
    __syncthreads();
    if (tid < 16) {
        float M = NEG_INF, S = 0.f;
        for (int w = 0; w < 8; ++w) {
            float m = smax[w][tid], s = ssum[w][tid];
            if (m > NEG_INF) {
                float nm = fmaxf(M, m);
                S = S * __expf(M - nm) + s * __expf(m - nm);
                M = nm;
            }
        }
        int grow = rowBase + tid;
        chunkMax[grow * NCHUNK + chunk] = M;
        chunkSum[grow * NCHUNK + chunk] = S;
    }
}

// =====================================================================
// Kernel 3: per-row merge of chunk stats -> log-partition; NLL at target;
// atomic mean into the loss slot. One thread per row.
// =====================================================================
__global__ __launch_bounds__(256)
void finalize_loss_kernel(const int* __restrict__ targets,
                          const float* __restrict__ logits,
                          const float* __restrict__ chunkMax,
                          const float* __restrict__ chunkSum,
                          float* __restrict__ loss)
{
    int row = blockIdx.x * blockDim.x + threadIdx.x;
    if (row >= ROWS) return;
    float M = NEG_INF, S = 0.f;
    for (int c = 0; c < NCHUNK; ++c) {
        float m = chunkMax[row * NCHUNK + c];
        float s = chunkSum[row * NCHUNK + c];
        if (m > NEG_INF) {
            float nm = fmaxf(M, m);
            S = S * __expf(M - nm) + s * __expf(m - nm);
            M = nm;
        }
    }
    int t = targets[row];
    float lt = logits[(size_t)row * VOCAB + t];
    float logp = lt - M - __logf(S);
    atomicAdd(loss, -logp * (1.0f / (float)ROWS));
}

// =====================================================================
extern "C" void kernel_launch(void* const* d_in, const int* in_sizes, int n_in,
                              void* d_out, int out_size, void* d_ws, size_t ws_size,
                              hipStream_t stream)
{
    const int*   idx     = (const int*)d_in[0];
    const int*   targets = (const int*)d_in[1];
    const float* tok_emb = (const float*)d_in[2];
    const float* pos_emb = (const float*)d_in[3];
    const float* Wq      = (const float*)d_in[4];
    const float* Wk      = (const float*)d_in[5];
    const float* Wv      = (const float*)d_in[6];
    const float* lm_W    = (const float*)d_in[7];
    const float* lm_b    = (const float*)d_in[8];

    float* logits = (float*)d_out;                       // [4096][50257]
    float* loss   = logits + (size_t)ROWS * VOCAB;       // scalar, return order #2

    // ---- workspace layout (~8.7 MB) ----
    unsigned short* xpack = (unsigned short*)d_ws;             // 256*2*32*16  = 512 KB
    unsigned short* wpack = xpack + (size_t)256 * 2 * 32 * FRAG_US;
    float* chunkMax = (float*)(wpack + (size_t)NTILE_B * 2 * 32 * FRAG_US); // 6.4 MB after
    float* chunkSum = chunkMax + (size_t)ROWS * NCHUNK;

    embed_attn_pack_kernel<<<256, 256, 0, stream>>>(idx, tok_emb, pos_emb,
                                                    Wq, Wk, Wv, xpack, loss);
    pack_lmW_kernel<<<(NTILE_B * 64 + 255) / 256, 256, 0, stream>>>(lm_W, wpack);
    gemm_logits_kernel<<<dim3(NCHUNK, ROWS / 16), 256, 0, stream>>>(
        xpack, wpack, lm_b, logits, chunkMax, chunkSum);
    finalize_loss_kernel<<<ROWS / 256, 256, 0, stream>>>(
        targets, logits, chunkMax, chunkSum, loss);
}
